// SwinTransformer_85761906966604
// MI455X (gfx1250) — matrix operand order repaired and checked by reading, not compiled, not run on hardware
//
#include <hip/hip_runtime.h>

typedef _Float16 h16;
typedef __attribute__((ext_vector_type(16))) _Float16 v16h;
typedef __attribute__((ext_vector_type(8)))  _Float16 v8h;
typedef __attribute__((ext_vector_type(4)))  _Float16 v4h;
typedef __attribute__((ext_vector_type(8)))  float    v8f;

#define NWIN   2048      // total windows (B=8 x 256)
#define NPAD   112       // 98 tokens padded to 7*16
#define NTOK   98
#define CDIM   128
#define HIDDIM 512
#define HD     32
#define THW    25088     // 8*56*56
#define TOKENS 200704    // 8*THW
#define QK_SCALE 0.17677669529663689f

__device__ __forceinline__ float wave_sum(float v){
#pragma unroll
  for (int m = 16; m >= 1; m >>= 1) v += __shfl_xor(v, m, 32);
  return v;
}
__device__ __forceinline__ float wave_max(float v){
#pragma unroll
  for (int m = 16; m >= 1; m >>= 1) v = fmaxf(v, __shfl_xor(v, m, 32));
  return v;
}

// A fragment: 16x32 f16 tile (rows row..row+15, K = k0..k0+31) from K-contiguous
// row-major storage (contiguous k order: k = 16*(lane/16) + s, matching B).
__device__ __forceinline__ v16h load_a_frag(const h16* base, int row, int ldk, int k0){
  int lane = threadIdx.x & 31;
  int g = lane >> 4;
  const h16* p = base + (size_t)(row + (lane & 15)) * ldk + k0 + g * 16;
  v8h lo = *(const v8h*)p;
  v8h hi = *(const v8h*)(p + 8);
  v16h r;
#pragma unroll
  for (int i = 0; i < 8; ++i){ r[i] = lo[i]; r[i + 8] = hi[i]; }
  return r;
}
// B fragment: 32x16 tile loaded from B^T stored row-major [N][K] (K-contiguous).
__device__ __forceinline__ v16h load_bt_frag(const h16* base, int col, int ldk, int k0){
  int lane = threadIdx.x & 31;
  int g = lane >> 4;
  const h16* p = base + (size_t)(col + (lane & 15)) * ldk + k0 + g * 16;
  v8h lo = *(const v8h*)p;
  v8h hi = *(const v8h*)(p + 8);
  v16h r;
#pragma unroll
  for (int i = 0; i < 8; ++i){ r[i] = lo[i]; r[i + 8] = hi[i]; }
  return r;
}

#define WMMA_F16(a,b,c) __builtin_amdgcn_wmma_f32_16x16x32_f16(false,(a),false,(b),(short)0,(c),false,false)

__device__ __forceinline__ int region3(int t, int h, int w){
  int rt = (t < 6) ? 0 : ((t < 7) ? 1 : 2);
  int rh = (h < 49) ? 0 : ((h < 53) ? 1 : 2);
  int rw = (w < 49) ? 0 : ((w < 53) ? 1 : 2);
  return rt * 9 + rh * 3 + rw;
}

// prefetch a weight panel (bytes) into cache: one global_prefetch_b8 per 256B line slice
__device__ __forceinline__ void prefetch_panel(const void* p, int bytes){
  int i = threadIdx.x;
  int lines = bytes >> 8;
  if (i < lines) __builtin_prefetch((const char*)p + i * 256, 0, 3);
}

// ---------------- weight transpose to f16 (src [R][C] -> dst [C][R]) ----------------
__global__ void k_transpose_f16(const float* __restrict__ src, h16* __restrict__ dst, int R, int C){
  int i = blockIdx.x * 256 + threadIdx.x;
  if (i >= R * C) return;
  int r = i / C, c = i - r * C;
  dst[(size_t)c * R + r] = (h16)src[i];
}

// ---- precompute bias+mask table T[8 classes][4 heads][112 nq][112 mk] (f32) ----
// class = (wt==3)<<2 | (wh==7)<<1 | (ww==7); padded key cols carry -1e9 sentinel.
__global__ void k_bias_table(const float* __restrict__ rpb, float* __restrict__ T){
  int i = blockIdx.x * 256 + threadIdx.x;          // 8*4*112*112 = 401408
  if (i >= 8 * 4 * 112 * 112) return;
  int mk = i % 112; int t_ = i / 112;
  int nq = t_ % 112; t_ /= 112;
  int head = t_ & 3; int cls = t_ >> 2;
  float v;
  if (mk >= NTOK){ v = -1e9f; }
  else {
    int nqc = nq < NTOK ? nq : NTOK - 1;
    int t1 = nqc / 49, r1 = nqc - t1 * 49, h1 = r1 / 7, w1 = r1 - h1 * 7;
    int t2 = mk  / 49, r2 = mk  - t2 * 49, h2 = r2 / 7, w2 = r2 - h2 * 7;
    int ridx = (t1 - t2 + 1) * 169 + (h1 - h2 + 6) * 13 + (w1 - w2 + 6);
    v = rpb[ridx * 4 + head];
    int wt = (cls & 4) ? 3 : 0, wh = (cls & 2) ? 7 : 0, ww = (cls & 1) ? 7 : 0;
    int ra = region3(wt * 2 + t1, wh * 7 + h1, ww * 7 + w1);
    int rb = region3(wt * 2 + t2, wh * 7 + h2, ww * 7 + w2);
    if (ra != rb) v -= 100.f;
  }
  T[i] = v;
}

// --------- LN1 + cyclic shift + window partition -> XW f16 [2048][112][128] ---------
// also emits POS[win][n] = flattened source/destination position (inverse-roll map)
__global__ __launch_bounds__(256) void k_ln1_window(const float* __restrict__ x,
    const float* __restrict__ g, const float* __restrict__ bvec, h16* __restrict__ xw,
    int* __restrict__ posw){
  int row  = blockIdx.x * 8 + (threadIdx.x >> 5);
  int lane = threadIdx.x & 31;
  int win = row / NPAD;
  int n   = row - win * NPAD;
  h16* dst = xw + (size_t)row * CDIM + lane * 4;
  if (n >= NTOK){ v4h z = {(h16)0.f,(h16)0.f,(h16)0.f,(h16)0.f}; *(v4h*)dst = z; return; }
  int bi = win >> 8, wt = (win >> 6) & 3, wh = (win >> 3) & 7, ww = win & 7;
  int t1 = n / 49; int rr = n - t1 * 49; int h1 = rr / 7; int w1 = rr - h1 * 7;
  int t = (wt * 2 + t1 + 1) & 7;
  int h = wh * 7 + h1 + 3; if (h >= 56) h -= 56;
  int w = ww * 7 + w1 + 3; if (w >= 56) w -= 56;
  int pos = bi * THW + t * 3136 + h * 56 + w;
  if (lane == 0) posw[win * NTOK + n] = pos;
  size_t idx = (size_t)pos * CDIM;
  float4 xv = *(const float4*)(x + idx + lane * 4);
  float mu = wave_sum(xv.x + xv.y + xv.z + xv.w) * (1.f / 128.f);
  float d0 = xv.x - mu, d1 = xv.y - mu, d2 = xv.z - mu, d3 = xv.w - mu;
  float var = wave_sum(d0*d0 + d1*d1 + d2*d2 + d3*d3) * (1.f / 128.f);
  float rstd = rsqrtf(var + 1e-5f);
  float4 gv = *(const float4*)(g + lane * 4);
  float4 bv = *(const float4*)(bvec + lane * 4);
  v4h o;
  o[0] = (h16)(d0 * rstd * gv.x + bv.x);
  o[1] = (h16)(d1 * rstd * gv.y + bv.y);
  o[2] = (h16)(d2 * rstd * gv.z + bv.z);
  o[3] = (h16)(d3 * rstd * gv.w + bv.w);
  *(v4h*)dst = o;
}

// ---- QKV GEMM, strip-mined: wave computes a 16x64 strip, A frags hoisted over kt ----
__global__ __launch_bounds__(256) void k_qkv(const h16* __restrict__ xw, const h16* __restrict__ wqkvT,
    const float* __restrict__ qkv_b, h16* __restrict__ Q, h16* __restrict__ K, h16* __restrict__ Vt){
  int win = blockIdx.x;
  int wid = threadIdx.x >> 5, lane = threadIdx.x & 31;
  int nlane = lane & 15, mhi = (lane >> 4) * 8;
  prefetch_panel(wqkvT, 384 * 128 * 2);
  const h16* abase = xw + (size_t)win * NPAD * CDIM;
  for (int task = wid; task < 42; task += 8){
    int mt = task / 6, ng = task - mt * 6;
    v16h a[4];
#pragma unroll
    for (int kt = 0; kt < 4; ++kt) a[kt] = load_a_frag(abase, mt * 16, CDIM, kt * 32);
#pragma unroll
    for (int j = 0; j < 4; ++j){
      int nt = ng * 4 + j;
      v8f acc = {0.f,0.f,0.f,0.f,0.f,0.f,0.f,0.f};
#pragma unroll
      for (int kt = 0; kt < 4; ++kt){
        v16h b = load_bt_frag(wqkvT, nt * 16, CDIM, kt * 32);
        acc = WMMA_F16(a[kt], b, acc);
      }
      float bias = qkv_b[nt * 16 + nlane];
      int part = nt >> 3;                 // 0=q 1=k 2=v  (tile-uniform)
      int head = (nt & 7) >> 1;
      int d    = ((nt & 1) << 4) + nlane;
      size_t qk = ((size_t)(win * 4 + head) * NPAD + mt * 16 + mhi) * HD + d;
      if (part == 0){
#pragma unroll
        for (int r = 0; r < 8; ++r) Q[qk + (size_t)r * HD] = (h16)((acc[r] + bias) * QK_SCALE);
      } else if (part == 1){
#pragma unroll
        for (int r = 0; r < 8; ++r) K[qk + (size_t)r * HD] = (h16)(acc[r] + bias);
      } else {
        size_t vo = ((size_t)(win * 4 + head) * HD + d) * 128 + mt * 16 + mhi;
#pragma unroll
        for (int r = 0; r < 8; ++r) Vt[vo + r] = (h16)(acc[r] + bias);
      }
    }
  }
}

// zero the padded token columns 112..127 of V^T
__global__ void k_vt_tail(h16* __restrict__ Vt){
  int i = blockIdx.x * 256 + threadIdx.x;          // 2048*4*32*16 elems
  int rowv = i >> 4, off = i & 15;
  Vt[(size_t)rowv * 128 + 112 + off] = (h16)0.f;
}

// -------- attention: one block per (window, head); S f32 + P f16 share static LDS --------
__global__ __launch_bounds__(256) void k_attn(const h16* __restrict__ Q, const h16* __restrict__ K,
    const h16* __restrict__ Vt, const float* __restrict__ T, h16* __restrict__ AO){
  __shared__ float S[112 * 116];                  // 51968 B; P aliases it with stride 232 halfs
  h16* P = (h16*)S;
  int win = blockIdx.x >> 2, head = blockIdx.x & 3;
  int wid = threadIdx.x >> 5, lane = threadIdx.x & 31;
  int wt = (win >> 6) & 3, wh = (win >> 3) & 7, ww = win & 7;
  int cls = ((wt == 3) << 2) | ((wh == 7) << 1) | (ww == 7);
  const float* Tb = T + (size_t)(cls * 4 + head) * 112 * 112;
  const h16* qb = Q + (size_t)(win * 4 + head) * NPAD * HD;
  const h16* kb = K + (size_t)(win * 4 + head) * NPAD * HD;
  int mhi = (lane >> 4) * 8;
  int nlane = lane & 15;

  // phase 1: S = Q@K^T + (precomputed bias+mask).  Wave owns an mt strip.
  for (int mt = wid; mt < 7; mt += 8){
    v16h a = load_a_frag(qb, mt * 16, HD, 0);
#pragma unroll
    for (int nt = 0; nt < 7; ++nt){
      v16h b = load_bt_frag(kb, nt * 16, HD, 0);
      v8f acc = {0.f,0.f,0.f,0.f,0.f,0.f,0.f,0.f};
      acc = WMMA_F16(a, b, acc);
      int mk = nt * 16 + nlane;
      int nq0 = mt * 16 + mhi;
#pragma unroll
      for (int r = 0; r < 8; ++r){
        int nq = nq0 + r;
        S[nq * 116 + mk] = acc[r] + Tb[nq * 112 + mk];
      }
    }
  }
  __syncthreads();

  // phase 2: softmax rows (each wave owns 14 rows)
  float pv[14][4];
  int k = 0;
  for (int row = wid; row < 112; row += 8, ++k){
    const float* sr = S + row * 116;
    float v0 = sr[lane], v1 = sr[lane + 32], v2 = sr[lane + 64];
    int c3 = lane + 96;
    float v3 = (c3 < 112) ? sr[c3] : -1e9f;
    float m = wave_max(fmaxf(fmaxf(v0, v1), fmaxf(v2, v3)));
    float e0 = expf(v0 - m), e1 = expf(v1 - m), e2 = expf(v2 - m), e3 = expf(v3 - m);
    float inv = 1.f / wave_sum(e0 + e1 + e2 + e3);
    pv[k][0] = e0 * inv; pv[k][1] = e1 * inv; pv[k][2] = e2 * inv; pv[k][3] = e3 * inv;
  }
  __syncthreads();                                  // all S reads done before P overwrite
  k = 0;
  for (int row = wid; row < 112; row += 8, ++k){
    h16* pr = P + row * 232;
    pr[lane]      = (h16)pv[k][0];
    pr[lane + 32] = (h16)pv[k][1];
    pr[lane + 64] = (h16)pv[k][2];
    int c3 = lane + 96;
    pr[c3] = (c3 < 112) ? (h16)pv[k][3] : (h16)0.f;  // zero padded K-cols 112..127
  }
  __syncthreads();

  // phase 3: out = P @ V.  Wave owns an mt strip: 4 A frags hoisted, reused over d-tiles.
  const h16* vb = Vt + (size_t)(win * 4 + head) * HD * 128;
  for (int mt = wid; mt < 7; mt += 8){
    v16h a[4];
#pragma unroll
    for (int kt = 0; kt < 4; ++kt) a[kt] = load_a_frag((const h16*)S, mt * 16, 232, kt * 32);
#pragma unroll
    for (int nt = 0; nt < 2; ++nt){
      v8f acc = {0.f,0.f,0.f,0.f,0.f,0.f,0.f,0.f};
#pragma unroll
      for (int kt = 0; kt < 4; ++kt){
        v16h b = load_bt_frag(vb, nt * 16, 128, kt * 32);
        acc = WMMA_F16(a[kt], b, acc);
      }
      size_t o = ((size_t)win * NPAD + mt * 16 + mhi) * CDIM + head * 32 + nt * 16 + nlane;
#pragma unroll
      for (int r = 0; r < 8; ++r) AO[o + (size_t)r * CDIM] = (h16)acc[r];
    }
  }
}

// -------- proj GEMM + window-reverse scatter (table-driven) + residual -> XRES f32 --------
__global__ __launch_bounds__(256) void k_proj(const h16* __restrict__ AO, const h16* __restrict__ wpT,
    const float* __restrict__ proj_b, const float* __restrict__ x, const int* __restrict__ posw,
    float* __restrict__ xres){
  int win = blockIdx.x;
  int wid = threadIdx.x >> 5, lane = threadIdx.x & 31;
  const h16* abase = AO + (size_t)win * NPAD * CDIM;
  const int* pw = posw + win * NTOK;
  int mhi = (lane >> 4) * 8, nlane = lane & 15;
  prefetch_panel(wpT, 128 * 128 * 2);
  for (int mt = wid; mt < 7; mt += 8){              // wave owns a 16-row strip
    v16h a[4];
#pragma unroll
    for (int kt = 0; kt < 4; ++kt) a[kt] = load_a_frag(abase, mt * 16, CDIM, kt * 32);
#pragma unroll
    for (int nt = 0; nt < 8; ++nt){
      v8f acc = {0.f,0.f,0.f,0.f,0.f,0.f,0.f,0.f};
#pragma unroll
      for (int kt = 0; kt < 4; ++kt){
        v16h b = load_bt_frag(wpT, nt * 16, CDIM, kt * 32);
        acc = WMMA_F16(a[kt], b, acc);
      }
      int col = nt * 16 + nlane;
      float pb = proj_b[col];
#pragma unroll
      for (int r = 0; r < 8; ++r){
        int tok = mt * 16 + r + mhi;
        if (tok < NTOK){
          size_t idx = (size_t)pw[tok] * CDIM + col;
          xres[idx] = x[idx] + acc[r] + pb;
        }
      }
    }
  }
}

// ---------------- LN2 -> f16 pack [200704][128] ----------------
__global__ __launch_bounds__(256) void k_ln2(const float* __restrict__ xres, const float* __restrict__ g,
    const float* __restrict__ bvec, h16* __restrict__ ln2){
  int row = blockIdx.x * 8 + (threadIdx.x >> 5);
  int lane = threadIdx.x & 31;
  float4 xv = *(const float4*)(xres + (size_t)row * CDIM + lane * 4);
  float mu = wave_sum(xv.x + xv.y + xv.z + xv.w) * (1.f / 128.f);
  float d0 = xv.x - mu, d1 = xv.y - mu, d2 = xv.z - mu, d3 = xv.w - mu;
  float var = wave_sum(d0*d0 + d1*d1 + d2*d2 + d3*d3) * (1.f / 128.f);
  float rstd = rsqrtf(var + 1e-5f);
  float4 gv = *(const float4*)(g + lane * 4);
  float4 bv = *(const float4*)(bvec + lane * 4);
  v4h o;
  o[0] = (h16)(d0 * rstd * gv.x + bv.x);
  o[1] = (h16)(d1 * rstd * gv.y + bv.y);
  o[2] = (h16)(d2 * rstd * gv.z + bv.z);
  o[3] = (h16)(d3 * rstd * gv.w + bv.w);
  *(v4h*)(ln2 + (size_t)row * CDIM + lane * 4) = o;
}

// -------- fused MLP: fc1 -> GELU (in LDS) -> fc2 + residual -> d_out --------
__global__ __launch_bounds__(256) void k_mlp(const h16* __restrict__ ln2, const h16* __restrict__ fc1T,
    const float* __restrict__ fc1b, const h16* __restrict__ fc2T, const float* __restrict__ fc2b,
    const float* __restrict__ xres, float* __restrict__ out){
  __shared__ h16 H1[16 * HIDDIM];                  // 16 KB
  int row0 = blockIdx.x * 16;
  int wid = threadIdx.x >> 5, lane = threadIdx.x & 31;
  int mhi = (lane >> 4) * 8, nlane = lane & 15;
  prefetch_panel(fc1T, 512 * 128 * 2);
  const h16* abase = ln2 + (size_t)row0 * CDIM;
  v16h a[4];                                        // A frags shared by all 32 n-tiles
#pragma unroll
  for (int kt = 0; kt < 4; ++kt) a[kt] = load_a_frag(abase, 0, CDIM, kt * 32);
  for (int nt = wid; nt < 32; nt += 8){
    v8f acc = {0.f,0.f,0.f,0.f,0.f,0.f,0.f,0.f};
#pragma unroll
    for (int kt = 0; kt < 4; ++kt){
      v16h b = load_bt_frag(fc1T, nt * 16, CDIM, kt * 32);
      acc = WMMA_F16(a[kt], b, acc);
    }
    int col = nt * 16 + nlane;
    float fb = fc1b[col];
#pragma unroll
    for (int r = 0; r < 8; ++r){
      float v = acc[r] + fb;
      v = 0.5f * v * (1.f + erff(v * 0.70710678118654752f));   // exact GELU
      H1[(r + mhi) * HIDDIM + col] = (h16)v;
    }
  }
  prefetch_panel(fc2T, 128 * 512 * 2);
  __syncthreads();
  {
    int nt = wid;                                  // 8 waves, 8 output n-tiles
    v8f acc = {0.f,0.f,0.f,0.f,0.f,0.f,0.f,0.f};
#pragma unroll
    for (int kt = 0; kt < 16; ++kt){
      v16h af = load_a_frag(H1, 0, HIDDIM, kt * 32);
      v16h b  = load_bt_frag(fc2T, nt * 16, HIDDIM, kt * 32);
      acc = WMMA_F16(af, b, acc);
    }
    int col = nt * 16 + nlane;
    float fb = fc2b[col];
#pragma unroll
    for (int r = 0; r < 8; ++r){
      size_t idx = (size_t)(row0 + r + mhi) * CDIM + col;
      out[idx] = xres[idx] + acc[r] + fb;
    }
  }
}

extern "C" void kernel_launch(void* const* d_in, const int* in_sizes, int n_in,
                              void* d_out, int out_size, void* d_ws, size_t ws_size,
                              hipStream_t stream){
  (void)in_sizes; (void)n_in; (void)out_size; (void)ws_size;
  const float* x      = (const float*)d_in[0];
  const float* n1g    = (const float*)d_in[1];
  const float* n1b    = (const float*)d_in[2];
  const float* qkv_w  = (const float*)d_in[3];
  const float* qkv_b  = (const float*)d_in[4];
  const float* proj_w = (const float*)d_in[5];
  const float* proj_b = (const float*)d_in[6];
  const float* rpb    = (const float*)d_in[7];
  const float* n2g    = (const float*)d_in[8];
  const float* n2b    = (const float*)d_in[9];
  const float* fc1_w  = (const float*)d_in[10];
  const float* fc1_b  = (const float*)d_in[11];
  const float* fc2_w  = (const float*)d_in[12];
  const float* fc2_b  = (const float*)d_in[13];
  float* out = (float*)d_out;

  char* ws = (char*)d_ws;
  size_t off = 0;
  auto take = [&](size_t bytes)->char*{
    char* p = ws + off; off += (bytes + 255) & ~(size_t)255; return p;
  };
  const size_t SZ_XW = (size_t)NWIN * NPAD * CDIM * 2;        // 58.7 MB
  const size_t SZ_Q  = (size_t)NWIN * 4 * NPAD * HD * 2;      // 58.7 MB
  const size_t SZ_K  = SZ_Q;
  const size_t SZ_VT = (size_t)NWIN * 4 * HD * 128 * 2;       // 67.1 MB
  h16* XW = (h16*)take(SZ_XW);          // later reused as AO (attn output)
  h16* Qb = (h16*)take(SZ_Q);           // later reused as LN2 pack
  h16* Kb = (h16*)take(SZ_K);           // Kb+Vt region later reused as XRES (f32)
  h16* Vt = (h16*)take(SZ_VT);
  h16* WQKVT = (h16*)take((size_t)384 * 128 * 2);
  h16* WPT   = (h16*)take((size_t)128 * 128 * 2);
  h16* F1T   = (h16*)take((size_t)512 * 128 * 2);
  h16* F2T   = (h16*)take((size_t)128 * 512 * 2);
  float* TBIAS = (float*)take((size_t)8 * 4 * 112 * 112 * 4); // 1.6 MB
  int*   POSW  = (int*)take((size_t)NWIN * NTOK * 4);         // 0.8 MB
  h16*   AO   = XW;                     // alias: XW dead after k_qkv
  h16*   LN2b = Qb;                     // alias: Q dead after k_attn
  float* XR   = (float*)Kb;             // alias: K,Vt dead after k_attn (125.8MB >= 102.8MB)

  k_transpose_f16<<<192, 256, 0, stream>>>(qkv_w, WQKVT, 128, 384);
  k_transpose_f16<<<64,  256, 0, stream>>>(proj_w, WPT, 128, 128);
  k_transpose_f16<<<256, 256, 0, stream>>>(fc1_w, F1T, 128, 512);
  k_transpose_f16<<<256, 256, 0, stream>>>(fc2_w, F2T, 512, 128);
  k_bias_table<<<1568, 256, 0, stream>>>(rpb, TBIAS);

  k_ln1_window<<<28672, 256, 0, stream>>>(x, n1g, n1b, XW, POSW);
  k_qkv<<<NWIN, 256, 0, stream>>>(XW, WQKVT, qkv_b, Qb, Kb, Vt);
  k_vt_tail<<<16384, 256, 0, stream>>>(Vt);
  k_attn<<<NWIN * 4, 256, 0, stream>>>(Qb, Kb, Vt, TBIAS, AO);
  k_proj<<<NWIN, 256, 0, stream>>>(AO, WPT, proj_b, x, POSW, XR);
  k_ln2<<<25088, 256, 0, stream>>>(XR, n2g, n2b, LN2b);
  k_mlp<<<12544, 256, 0, stream>>>(LN2b, F1T, fc1_b, F2T, fc2_b, XR, out);
}
